// Stance_77567109366329
// MI455X (gfx1250) — matrix operand order repaired and checked by reading, not compiled, not run on hardware
//
#include <hip/hip_runtime.h>
#include <hip/hip_bf16.h>

typedef __attribute__((ext_vector_type(16))) _Float16 v16h;
typedef __attribute__((ext_vector_type(8)))  _Float16 v8h;
typedef __attribute__((ext_vector_type(8)))  float    v8f;

// ---------------- problem constants ----------------
#define B_   256
#define L_   100
#define H_   512
#define EPS_ 1.0f
#define NITER_ 100

// padded NHWC activation geometry: left/top halo 3, right halo covers tile overrun
#define PH 106   // 3 + 100 + 3
#define PW 120   // 3 + 100 + 16 (+1 align)

// ---------------- workspace layout (bytes) ----------------
static constexpr size_t SIM_OFF  = 0;
static constexpr size_t MAXP_OFF = 10240000;
static constexpr size_t MAXS_OFF = MAXP_OFF + 2048u * 4u;          // 10,248,192
static constexpr size_t WT2_OFF  = 10248448;
static constexpr size_t WT3_OFF  = WT2_OFF + 51200;                // 10,299,648
static constexpr size_t WT4_OFF  = WT3_OFF + 102400;               // 10,402,048
static constexpr size_t ACT1_OFF = WT4_OFF + 401408;               // 10,803,456
static constexpr size_t ACT1_BYTES = (size_t)B_ * PH * PW * 32 * 2; // 208,404,480
static constexpr size_t ACT2_OFF = ACT1_OFF + ACT1_BYTES;
static constexpr size_t ACT3_OFF = ACT2_OFF + ACT1_BYTES;
static constexpr size_t ACT3_BYTES = (size_t)B_ * PH * PW * 64 * 2;
static constexpr size_t ACT4_OFF = ACT1_OFF;                        // alias (act1/act2 dead by then)

// =====================================================================
// 1) batched GEMM  sim[b,q,c] = sum_h qry[b,q,h]*cnd[b,c,h]  via WMMA f16
// =====================================================================
__global__ void __launch_bounds__(32)
gemm_sim_kernel(const float* __restrict__ qry, const float* __restrict__ cnd,
                float* __restrict__ sim) {
  const int tx = blockIdx.x;          // N tile (cnd index)
  const int ty = blockIdx.y;          // M tile (qry index)
  const int b  = blockIdx.z;
  const int lane = threadIdx.x;
  const int m  = lane & 15;
  const int q  = ty * 16 + m;
  const int n  = tx * 16 + m;
  const int kbA = (lane < 16) ? 0 : 8;
  const int kbB = (lane < 16) ? 0 : 16;
  const float qs = (q < L_) ? 1.0f : 0.0f;
  const float ns = (n < L_) ? 1.0f : 0.0f;
  const float* qrow = qry + ((size_t)b * L_ + (q < L_ ? q : 0)) * H_;
  const float* crow = cnd + ((size_t)b * L_ + (n < L_ ? n : 0)) * H_;

  v8f acc;
#pragma unroll
  for (int r = 0; r < 8; ++r) acc[r] = 0.0f;

  for (int k0 = 0; k0 < H_; k0 += 32) {
    const float4* qp0 = (const float4*)(qrow + k0 + kbB);
    const float4* qp1 = (const float4*)(qrow + k0 + 8 + kbB);
    float4 a0 = qp0[0], a1 = qp0[1];
    float4 a2 = qp1[0], a3 = qp1[1];
    v16h a;
    a[0]  = (_Float16)(a0.x * qs); a[1]  = (_Float16)(a0.y * qs);
    a[2]  = (_Float16)(a0.z * qs); a[3]  = (_Float16)(a0.w * qs);
    a[4]  = (_Float16)(a1.x * qs); a[5]  = (_Float16)(a1.y * qs);
    a[6]  = (_Float16)(a1.z * qs); a[7]  = (_Float16)(a1.w * qs);
    a[8]  = (_Float16)(a2.x * qs); a[9]  = (_Float16)(a2.y * qs);
    a[10] = (_Float16)(a2.z * qs); a[11] = (_Float16)(a2.w * qs);
    a[12] = (_Float16)(a3.x * qs); a[13] = (_Float16)(a3.y * qs);
    a[14] = (_Float16)(a3.z * qs); a[15] = (_Float16)(a3.w * qs);

    const float4* cp = (const float4*)(crow + k0 + kbB);
    float4 b0 = cp[0], b1 = cp[1], b2 = cp[2], b3 = cp[3];
    v16h bb;
    bb[0]  = (_Float16)(b0.x * ns); bb[1]  = (_Float16)(b0.y * ns);
    bb[2]  = (_Float16)(b0.z * ns); bb[3]  = (_Float16)(b0.w * ns);
    bb[4]  = (_Float16)(b1.x * ns); bb[5]  = (_Float16)(b1.y * ns);
    bb[6]  = (_Float16)(b1.z * ns); bb[7]  = (_Float16)(b1.w * ns);
    bb[8]  = (_Float16)(b2.x * ns); bb[9]  = (_Float16)(b2.y * ns);
    bb[10] = (_Float16)(b2.z * ns); bb[11] = (_Float16)(b2.w * ns);
    bb[12] = (_Float16)(b3.x * ns); bb[13] = (_Float16)(b3.y * ns);
    bb[14] = (_Float16)(b3.z * ns); bb[15] = (_Float16)(b3.w * ns);

    acc = __builtin_amdgcn_wmma_f32_16x16x32_f16(false, a, false, bb,
                                                 (short)0, acc, false, false);
  }

  const int mb = (lane < 16) ? 0 : 8;
  const int q0 = ty * 16 + mb;
  if (tx < 6 && ty < 6) {
    // full tile: 8 unconditional stores, single base address + static offsets
    float* op = sim + ((size_t)b * L_ + q0) * L_ + n;
#pragma unroll
    for (int r = 0; r < 8; ++r) op[(size_t)r * L_] = acc[r];
  } else {
#pragma unroll
    for (int r = 0; r < 8; ++r) {
      int qq = q0 + r;
      if (qq < L_ && n < L_)
        sim[((size_t)b * L_ + qq) * L_ + n] = acc[r];
    }
  }
}

// =====================================================================
// 2) global max over sim (two stages)
// =====================================================================
__global__ void reduce_max1(const float* __restrict__ x, int n, float* __restrict__ part) {
  __shared__ float sh[256];
  int t = threadIdx.x;
  float m = -INFINITY;
  for (int i = blockIdx.x * blockDim.x + t; i < n; i += gridDim.x * blockDim.x)
    m = fmaxf(m, x[i]);
  sh[t] = m; __syncthreads();
  for (int o = 128; o > 0; o >>= 1) { if (t < o) sh[t] = fmaxf(sh[t], sh[t + o]); __syncthreads(); }
  if (t == 0) part[blockIdx.x] = sh[0];
}

__global__ void reduce_max2(const float* __restrict__ part, int n, float* __restrict__ outmax) {
  __shared__ float sh[256];
  int t = threadIdx.x;
  float m = -INFINITY;
  for (int i = t; i < n; i += 256) m = fmaxf(m, part[i]);
  sh[t] = m; __syncthreads();
  for (int o = 128; o > 0; o >>= 1) { if (t < o) sh[t] = fmaxf(sh[t], sh[t + o]); __syncthreads(); }
  if (t == 0) *outmax = sh[0];
}

// =====================================================================
// 3) Sinkhorn: one block per batch, C matrix resident in LDS.
//    Overwrites simBuf with aligned = sim * pi * mask.
// =====================================================================
__global__ void __launch_bounds__(128)
sinkhorn_kernel(float* __restrict__ simBuf,
                const float* __restrict__ qmsk, const float* __restrict__ cmsk,
                const float* __restrict__ maxPtr) {
  __shared__ float S[L_][105];
  __shared__ float u[L_], v[L_], lmu[L_], lnu[L_], qm[L_], cm[L_];
  __shared__ float musum, nusum;
  const int b = blockIdx.x;
  const int t = threadIdx.x;
  const float mx1 = *maxPtr + 1e-6f;           // dist = mx1 - sim  (C)
  const float invE = 1.0f / EPS_;

  for (int i = t; i < L_ * L_; i += 128) S[i / L_][i % L_] = simBuf[(size_t)b * L_ * L_ + i];
  if (t < L_) { qm[t] = qmsk[b * L_ + t]; cm[t] = cmsk[b * L_ + t]; u[t] = 0.f; v[t] = 0.f; }
  __syncthreads();
  if (t == 0) {
    float s0 = 0.f, s1 = 0.f;
    for (int i = 0; i < L_; ++i) { s0 += qm[i] * cm[0]; s1 += qm[0] * cm[i]; }
    musum = s0; nusum = s1;
  }
  __syncthreads();
  if (t < L_) {
    lmu[t] = __logf(qm[t] * cm[0] / musum);
    lnu[t] = __logf(qm[0] * cm[t] / nusum);
  }
  __syncthreads();

  for (int it = 0; it < NITER_; ++it) {
    if (t < L_) {                               // row update: u
      float ut = u[t];
      float s = 0.f;
      for (int c = 0; c < L_; ++c)
        s += __expf((S[t][c] - mx1 + ut + v[c]) * invE);
      u[t] = EPS_ * (lmu[t] - __logf(s + 1e-6f)) + ut;
    }
    __syncthreads();
    if (t < L_) {                               // col update: v
      float vt = v[t];
      float s = 0.f;
      for (int q = 0; q < L_; ++q)
        s += __expf((S[q][t] - mx1 + u[q] + vt) * invE);
      v[t] = EPS_ * (lnu[t] - __logf(s + 1e-6f)) + vt;
    }
    __syncthreads();
  }

  for (int i = t; i < L_ * L_; i += 128) {
    int q = i / L_, c = i % L_;
    float s = S[q][c];
    float pi = __expf((s - mx1 + u[q] + v[c]) * invE);
    simBuf[(size_t)b * L_ * L_ + i] = s * pi * (qm[q] * cm[c]);
  }
}

// =====================================================================
// 4) weight transform: OIHW f32 -> (R,S,O,I) f16
// =====================================================================
__global__ void wxform_kernel(const float* __restrict__ src, _Float16* __restrict__ dst,
                              int O, int I, int R, int S) {
  int idx = blockIdx.x * 256 + threadIdx.x;
  if (idx >= O * I * R * S) return;
  int s = idx % S;
  int r = (idx / S) % R;
  int i = (idx / (S * R)) % I;
  int o = idx / (S * R * I);
  dst[(((size_t)(r * S + s) * O) + o) * I + i] = (_Float16)src[idx];
}

// =====================================================================
// 5) conv1: 1->32 ch, 3x3 pad1, + bias, ReLU, write padded NHWC f16
// =====================================================================
__global__ void __launch_bounds__(256)
conv1_kernel(const float* __restrict__ aligned, const float* __restrict__ w1,
             const float* __restrict__ b1, _Float16* __restrict__ act1) {
  __shared__ float wsh[32 * 9];
  __shared__ float bsh[32];
  int t = threadIdx.x;
  if (t < 288) wsh[t] = w1[t];
  if (t < 32)  bsh[t] = b1[t];
  __syncthreads();
  int id = blockIdx.x * 256 + t;
  if (id >= B_ * L_ * L_) return;
  int x = id % L_, y = (id / L_) % L_, b = id / (L_ * L_);
  float vin[9];
#pragma unroll
  for (int r = 0; r < 3; ++r)
#pragma unroll
    for (int s = 0; s < 3; ++s) {
      int yy = y + r - 1, xx = x + s - 1;
      vin[r * 3 + s] = (yy >= 0 && yy < L_ && xx >= 0 && xx < L_)
                       ? aligned[((size_t)b * L_ + yy) * L_ + xx] : 0.f;
    }
  _Float16* op = act1 + (((size_t)b * PH + y + 3) * PW + x + 3) * 32;
  for (int co = 0; co < 32; ++co) {
    float s = bsh[co];
#pragma unroll
    for (int j = 0; j < 9; ++j) s += vin[j] * wsh[co * 9 + j];
    op[co] = (_Float16)(s > 0.f ? s : 0.f);
  }
}

// =====================================================================
// 6) WMMA implicit-GEMM conv: one wave = 16 pixels (x) * 16 out-channels
//    Input NHWC f16 padded (halo >= PAD, zeros). Weights (R,S,COUT,CIN) f16.
// =====================================================================
template <int CIN, int COUT, int KS, int PAD, bool RELU, bool PADOUT>
__global__ void __launch_bounds__(32)
conv_wmma_kernel(const _Float16* __restrict__ in, const _Float16* __restrict__ wt,
                 const float* __restrict__ bias, _Float16* __restrict__ out) {
  const int xt = blockIdx.x;                 // 0..6
  const int y  = blockIdx.y;                 // 0..99
  const int zn = blockIdx.z;
  const int nt = zn % (COUT / 16);
  const int b  = zn / (COUT / 16);
  const int lane = threadIdx.x;
  const int m  = lane & 15;
  const int x0 = xt * 16;
  const int n0 = nt * 16;
  const int kbA = (lane < 16) ? 0 : 8;
  const int kbB = (lane < 16) ? 0 : 16;

  v8f acc;
  {
    float bv = bias[n0 + m];                 // column n = lane&15, same bias all rows
#pragma unroll
    for (int r = 0; r < 8; ++r) acc[r] = bv;
  }

  for (int r = 0; r < KS; ++r) {
    const int yin = y + r - PAD;
    const _Float16* inrow = in + ((size_t)b * PH + (yin + 3)) * PW * CIN;
    for (int s = 0; s < KS; ++s) {
      const int xin = x0 + m + s - PAD;                       // in [-PAD, 111+PAD], halo covers it
      const _Float16* ap = inrow + (size_t)(xin + 3) * CIN;
      const _Float16* wrow = wt + ((size_t)(r * KS + s) * COUT + n0 + m) * CIN;
#pragma unroll
      for (int cc = 0; cc < CIN; cc += 32) {
        v8h lo = *(const v8h*)(ap + cc + kbB);
        v8h hi = *(const v8h*)(ap + cc + kbB + 8);
        v16h afrag = __builtin_shufflevector(lo, hi, 0, 1, 2, 3, 4, 5, 6, 7,
                                             8, 9, 10, 11, 12, 13, 14, 15);
        v16h bfrag = *(const v16h*)(wrow + cc + kbB);
        acc = __builtin_amdgcn_wmma_f32_16x16x32_f16(false, afrag, false, bfrag,
                                                     (short)0, acc, false, false);
      }
    }
  }

  const int nn = n0 + m;
  const int mbase = (lane < 16) ? 0 : 8;
  const int xs = x0 + mbase;                 // first of the 8 output x's for this lane
  if (xt < 6) {
    // full tile (6/7 of waves): 8 unconditional b16 stores, static row offsets
    _Float16* op;
    if (PADOUT) op = out + (((size_t)b * PH + y + 3) * PW + xs + 3) * COUT + nn;
    else        op = out + (((size_t)b * L_ + y) * L_ + xs) * COUT + nn;
#pragma unroll
    for (int r = 0; r < 8; ++r) {
      float v = acc[r];
      if (RELU) v = v > 0.f ? v : 0.f;
      op[(size_t)r * COUT] = (_Float16)v;
    }
  } else {
#pragma unroll
    for (int r = 0; r < 8; ++r) {
      int xo = xs + r;
      if (xo < L_) {
        float v = acc[r];
        if (RELU) v = v > 0.f ? v : 0.f;
        size_t off;
        if (PADOUT) off = (((size_t)b * PH + y + 3) * PW + xo + 3) * COUT + nn;
        else        off = (((size_t)b * L_ + y) * L_ + xo) * COUT + nn;
        out[off] = (_Float16)v;
      }
    }
  }
}

// =====================================================================
// 7) maxpool(2x2) + 1x1 conv (wf,bf) + masked mean  -> out[b]
// =====================================================================
__global__ void __launch_bounds__(256)
final_kernel(const _Float16* __restrict__ act4, const float* __restrict__ wf,
             const float* __restrict__ bf, const float* __restrict__ qmsk,
             const float* __restrict__ cmsk, float* __restrict__ out) {
  __shared__ float wsh[64];
  __shared__ float racc[256], rpm[256];
  const int b = blockIdx.x;
  const int t = threadIdx.x;
  if (t < 64) wsh[t] = wf[t];
  __syncthreads();
  const float bfv = bf[0];
  float accs = 0.f, pms = 0.f;
  for (int p = t; p < 50 * 50; p += 256) {
    int py = p / 50, px = p % 50;
    const _Float16* base = act4 + (((size_t)b * L_ + 2 * py) * L_ + 2 * px) * 64;
    float val = bfv;
    for (int c = 0; c < 64; ++c) {
      float a0 = (float)base[c];
      float a1 = (float)base[64 + c];
      float a2 = (float)base[L_ * 64 + c];
      float a3 = (float)base[L_ * 64 + 64 + c];
      val += fmaxf(fmaxf(a0, a1), fmaxf(a2, a3)) * wsh[c];
    }
    float q0 = qmsk[b * L_ + 2 * py], q1 = qmsk[b * L_ + 2 * py + 1];
    float c0 = cmsk[b * L_ + 2 * px], c1 = cmsk[b * L_ + 2 * px + 1];
    float pm = fmaxf(fmaxf(q0 * c0, q0 * c1), fmaxf(q1 * c0, q1 * c1));
    accs += val * pm;
    pms  += pm;
  }
  racc[t] = accs; rpm[t] = pms; __syncthreads();
  for (int o = 128; o > 0; o >>= 1) {
    if (t < o) { racc[t] += racc[t + o]; rpm[t] += rpm[t + o]; }
    __syncthreads();
  }
  if (t == 0) out[b] = racc[0] / rpm[0];
}

// =====================================================================
extern "C" void kernel_launch(void* const* d_in, const int* in_sizes, int n_in,
                              void* d_out, int out_size, void* d_ws, size_t ws_size,
                              hipStream_t stream) {
  const float* qry  = (const float*)d_in[0];
  const float* cnd  = (const float*)d_in[1];
  const float* qmsk = (const float*)d_in[2];
  const float* cmsk = (const float*)d_in[3];
  const float* w1   = (const float*)d_in[4];
  const float* b1   = (const float*)d_in[5];
  const float* w2   = (const float*)d_in[6];
  const float* b2   = (const float*)d_in[7];
  const float* w3   = (const float*)d_in[8];
  const float* b3   = (const float*)d_in[9];
  const float* w4   = (const float*)d_in[10];
  const float* b4   = (const float*)d_in[11];
  const float* wf   = (const float*)d_in[12];
  const float* bf   = (const float*)d_in[13];
  float* outp = (float*)d_out;

  char* ws = (char*)d_ws;
  float*     simBuf = (float*)(ws + SIM_OFF);
  float*     parts  = (float*)(ws + MAXP_OFF);
  float*     maxs   = (float*)(ws + MAXS_OFF);
  _Float16*  wt2    = (_Float16*)(ws + WT2_OFF);
  _Float16*  wt3    = (_Float16*)(ws + WT3_OFF);
  _Float16*  wt4    = (_Float16*)(ws + WT4_OFF);
  _Float16*  act1   = (_Float16*)(ws + ACT1_OFF);
  _Float16*  act2   = (_Float16*)(ws + ACT2_OFF);
  _Float16*  act3   = (_Float16*)(ws + ACT3_OFF);
  _Float16*  act4   = (_Float16*)(ws + ACT4_OFF);

  // weight transforms (tiny)
  wxform_kernel<<<(32 * 32 * 5 * 5 + 255) / 256, 256, 0, stream>>>(w2, wt2, 32, 32, 5, 5);
  wxform_kernel<<<(64 * 32 * 5 * 5 + 255) / 256, 256, 0, stream>>>(w3, wt3, 64, 32, 5, 5);
  wxform_kernel<<<(64 * 64 * 7 * 7 + 255) / 256, 256, 0, stream>>>(w4, wt4, 64, 64, 7, 7);

  // sim = qry @ cnd^T (WMMA)
  gemm_sim_kernel<<<dim3(7, 7, B_), 32, 0, stream>>>(qry, cnd, simBuf);

  // global max(sim)
  reduce_max1<<<2048, 256, 0, stream>>>(simBuf, B_ * L_ * L_, parts);
  reduce_max2<<<1, 256, 0, stream>>>(parts, 2048, maxs);

  // Sinkhorn (writes aligned in place of sim)
  sinkhorn_kernel<<<B_, 128, 0, stream>>>(simBuf, qmsk, cmsk, maxs);

  // zero activation halos (interiors fully rewritten below)
  hipMemsetAsync(ws + ACT1_OFF, 0, ACT1_BYTES * 2, stream);   // act1 + act2 (contiguous)
  hipMemsetAsync(ws + ACT3_OFF, 0, ACT3_BYTES, stream);       // act3

  // conv1 (direct), ReLU fused
  conv1_kernel<<<(B_ * L_ * L_) / 256, 256, 0, stream>>>(simBuf, w1, b1, act1);

  // conv2/3/4 (WMMA implicit GEMM)
  conv_wmma_kernel<32, 32, 5, 2, true,  true ><<<dim3(7, L_, B_ * 2), 32, 0, stream>>>(act1, wt2, b2, act2);
  conv_wmma_kernel<32, 64, 5, 2, true,  true ><<<dim3(7, L_, B_ * 4), 32, 0, stream>>>(act2, wt3, b3, act3);
  conv_wmma_kernel<64, 64, 7, 3, false, false><<<dim3(7, L_, B_ * 4), 32, 0, stream>>>(act3, wt4, b4, act4);

  // maxpool + 1x1 conv + masked mean
  final_kernel<<<B_, 256, 0, stream>>>(act4, wf, bf, qmsk, cmsk, outp);
}
